// GlobalRouters_52450140618979
// MI455X (gfx1250) — matrix operand order repaired and checked by reading, not compiled, not run on hardware
//
#include <hip/hip_runtime.h>
#include <hip/hip_bf16.h>

typedef __attribute__((ext_vector_type(16))) _Float16 v16h;
typedef __attribute__((ext_vector_type(8)))  float    v8f;
typedef __attribute__((ext_vector_type(4)))  float    v4f;

#define B_      4
#define S_      2048
#define DM      1024
#define DS      64
#define NPOOL   2048
#define TOPK    8
#define MROWS   8192        // B_*S_
#define NSIX    384         // 6*DS
#define LDSTR   2052        // padded LDS row stride (floats)

// ---------------- wave32 reductions ----------------
__device__ __forceinline__ float wave_max32(float v) {
#pragma unroll
  for (int off = 16; off > 0; off >>= 1) v = fmaxf(v, __shfl_xor(v, off, 32));
  return v;
}
__device__ __forceinline__ float wave_sum32(float v) {
#pragma unroll
  for (int off = 16; off > 0; off >>= 1) v += __shfl_xor(v, off, 32);
  return v;
}

// A-fragment gather for v_wmma_f32_16x16x32_f16, element s holds K 16*(lane>=16) + s, as B does.
// rowbase already points at (row m, k0 + 16*(lane>=16)); contiguous K order.
__device__ __forceinline__ v16h load_a_frag(const _Float16* __restrict__ rowbase) {
  v16h a;
#pragma unroll
  for (int j = 0; j < 4; ++j) {          // VGPR 0..3 : K = {0..7} rel
    a[2*j]     = rowbase[2*j];
    a[2*j + 1] = rowbase[2*j + 1];
  }
#pragma unroll
  for (int j = 0; j < 4; ++j) {          // VGPR 4..7 : K = {8..15} rel
    a[8 + 2*j] = rowbase[8 + 2*j];
    a[9 + 2*j] = rowbase[8 + 2*j + 1];
  }
  return a;
}

// ---------------- prep kernels ----------------
__global__ void k_cvt_x(const float* __restrict__ x, _Float16* __restrict__ xf) {
  int i = blockIdx.x * blockDim.x + threadIdx.x;   // i indexes float4 chunks
  float4 v = ((const float4*)x)[i];
  _Float16* o = xf + 4 * (size_t)i;
  o[0] = (_Float16)v.x; o[1] = (_Float16)v.y;
  o[2] = (_Float16)v.z; o[3] = (_Float16)v.w;
}

__global__ void k_cvt_w(const float* __restrict__ W, _Float16* __restrict__ Wt) {
  int n = blockIdx.x;                     // 0..383
#pragma unroll
  for (int j = 0; j < 4; ++j) {
    int k = threadIdx.x + 256 * j;        // 0..1023
    Wt[n * DM + k] = (_Float16)W[k * NSIX + n];
  }
}

__global__ void k_norm_emb(const float* __restrict__ emb, _Float16* __restrict__ embf) {
  int lane = threadIdx.x & 31, w = threadIdx.x >> 5;
  int row = blockIdx.x * 8 + w;           // 0..8191 (pools 0..3 only)
  float e0 = emb[row * DS + lane];
  float e1 = emb[row * DS + 32 + lane];
  float ss = wave_sum32(e0 * e0 + e1 * e1);
  float inv = 1.0f / fmaxf(sqrtf(ss), 1e-12f);
  embf[row * DS + lane]      = (_Float16)(e0 * inv);
  embf[row * DS + 32 + lane] = (_Float16)(e1 * inv);
}

// ---------------- proj GEMM: proj[8192,384] = x_f16 @ W_f16 + b ----------------
__global__ __launch_bounds__(256) void k_proj(const _Float16* __restrict__ xf,
                                              const _Float16* __restrict__ Wt,
                                              const float* __restrict__ bias,
                                              _Float16* __restrict__ projf) {
  int lane = threadIdx.x & 31, w = threadIdx.x >> 5;
  int wid = blockIdx.x * 8 + w;           // 0..2047
  int mb = wid >> 2;                      // 0..511  (16-row block)
  int ns = wid & 3;                       // 0..3    (96-col super-tile)
  int ahalf = (lane >> 4) << 3;           // 0 / 8
  int bhalf = (lane >> 4) << 4;           // 0 / 16
  const _Float16* arow = xf + (size_t)(mb * 16 + (lane & 15)) * DM;
  const _Float16* brow = Wt + (size_t)(ns * 96 + (lane & 15)) * DM + bhalf;

  v8f acc[6] = {};
  for (int k0 = 0; k0 < DM; k0 += 32) {
    // issue the whole k-step's loads first, then drain into WMMAs
    v16h A = load_a_frag(arow + k0 + bhalf);
    v16h Bf[6];
#pragma unroll
    for (int t = 0; t < 6; ++t)
      Bf[t] = *(const v16h*)(brow + (size_t)(t * 16) * DM + k0);
#pragma unroll
    for (int t = 0; t < 6; ++t)
      acc[t] = __builtin_amdgcn_wmma_f32_16x16x32_f16(false, A, false, Bf[t],
                                                      (short)0, acc[t], false, false);
  }
#pragma unroll
  for (int t = 0; t < 6; ++t) {
    int n = ns * 96 + t * 16 + (lane & 15);
    float bv = bias[n];
#pragma unroll
    for (int r = 0; r < 8; ++r) {
      int m = mb * 16 + r + ahalf;        // C/D layout: M = r + 8*(lane>=16)
      projf[(size_t)m * NSIX + n] = (_Float16)(acc[t][r] + bv);
    }
  }
}

// ------- fused: logits (WMMA) -> softmax -> top-8 -> sparse dense write -------
__global__ __launch_bounds__(512) void k_router(const _Float16* __restrict__ projf,
                                                const _Float16* __restrict__ embf,
                                                float* __restrict__ out) {
  extern __shared__ float lds[];          // [16][LDSTR]
  const int head = blockIdx.y;            // 0..5
  const int mb   = blockIdx.x;            // 0..511
  const int lane = threadIdx.x & 31, w = threadIdx.x >> 5;   // 16 waves
  const int pool = (head < 2) ? 0 : (head == 2) ? 1 : (head < 5) ? 2 : 3;
  const int cbase = w * 128;              // this wave's 128 columns
  const int ahalf = (lane >> 4) << 3;
  const int bhalf = (lane >> 4) << 4;
  const _Float16* arow = projf + (size_t)(mb * 16 + (lane & 15)) * NSIX + head * DS;
  const _Float16* brow = embf + (size_t)(pool * NPOOL + cbase + (lane & 15)) * DS + bhalf;
  __builtin_prefetch(arow, 0, 0);         // global_prefetch_b8

  v8f acc[8] = {};
#pragma unroll
  for (int k0 = 0; k0 < DS; k0 += 32) {
    v16h A = load_a_frag(arow + k0 + bhalf);
    v16h Bf[8];
#pragma unroll
    for (int t = 0; t < 8; ++t)
      Bf[t] = *(const v16h*)(brow + (size_t)(t * 16) * DS + k0);
#pragma unroll
    for (int t = 0; t < 8; ++t)
      acc[t] = __builtin_amdgcn_wmma_f32_16x16x32_f16(false, A, false, Bf[t],
                                                      (short)0, acc[t], false, false);
  }
  // spill logits block to LDS: rows 0..15, cols cbase..cbase+127
#pragma unroll
  for (int t = 0; t < 8; ++t) {
    int c = cbase + t * 16 + (lane & 15);
#pragma unroll
    for (int r = 0; r < 8; ++r) lds[(r + ahalf) * LDSTR + c] = acc[t][r];
  }
  __syncthreads();

  // phase 2: wave w owns row w (2048 logits, 64 per lane, stride-32 = bank-clean)
  float* row = lds + w * LDSTR;
  float rmax = -3.402823466e38f;
  for (int i = 0; i < 64; ++i) rmax = fmaxf(rmax, row[i * 32 + lane]);
  rmax = wave_max32(rmax);

  float z = 0.0f;
  for (int i = 0; i < 64; ++i) z += __expf(row[i * 32 + lane] - rmax);
  z = wave_sum32(z);

  int   selIdx[TOPK];
  float selVal[TOPK];
  float s8 = 0.0f;
  for (int p = 0; p < TOPK; ++p) {
    float bv = -3.402823466e38f;
    int   bi = 0x7fffffff;
    for (int i = 0; i < 64; ++i) {
      float v = row[i * 32 + lane];
      if (v > bv) { bv = v; bi = i * 32 + lane; }    // keeps lowest index on ties
    }
#pragma unroll
    for (int off = 16; off > 0; off >>= 1) {         // wave argmax, tie -> low idx
      float ov = __shfl_xor(bv, off, 32);
      int   oi = __shfl_xor(bi, off, 32);
      if (ov > bv || (ov == bv && oi < bi)) { bv = ov; bi = oi; }
    }
    selIdx[p] = bi; selVal[p] = bv;
    s8 += __expf(bv - rmax);
    if (lane == 0) row[bi] = -3.402823466e38f;       // knock out winner
  }
  // final_i = exp(l_i - m) / (sum_top8 exp + 1e-8 * Z)
  float invden = 1.0f / (s8 + 1e-8f * z);

  size_t obase = ((size_t)(head * MROWS + mb * 16 + w)) * (size_t)NPOOL;
  // dense zero row: non-temporal streaming stores (402 MB total > 192 MB L2 --
  // keep L2 for the 25 MB resident working set)
  v4f* o4 = (v4f*)(out + obase);
  v4f zero4 = {0.f, 0.f, 0.f, 0.f};
#pragma unroll
  for (int i = 0; i < 16; ++i)
    __builtin_nontemporal_store(zero4, &o4[i * 32 + lane]);
  asm volatile("s_wait_storecnt 0x0" ::: "memory");  // zeros before scatter
  if (lane == 0) {
#pragma unroll
    for (int p = 0; p < TOPK; ++p)
      out[obase + (size_t)selIdx[p]] = __expf(selVal[p] - rmax) * invden;
  }
}

// ---------------- host launcher ----------------
extern "C" void kernel_launch(void* const* d_in, const int* in_sizes, int n_in,
                              void* d_out, int out_size, void* d_ws, size_t ws_size,
                              hipStream_t stream) {
  const float* x   = (const float*)d_in[0];   // (4,2048,1024)
  const float* W   = (const float*)d_in[1];   // (1024,384)
  const float* bia = (const float*)d_in[2];   // (384,)
  const float* emb = (const float*)d_in[3];   // (10240,64)
  float* out = (float*)d_out;                 // (6,4,2048,2048)

  char* ws = (char*)d_ws;
  _Float16* xf    = (_Float16*)(ws);                          // 16,777,216 B
  _Float16* Wt    = (_Float16*)(ws + 16777216);               //    786,432 B
  _Float16* embf  = (_Float16*)(ws + 16777216 + 786432);      //  1,048,576 B
  _Float16* projf = (_Float16*)(ws + 16777216 + 786432 + 1048576); // 6,291,456 B

  // prep: x -> f16 (float4 chunks), W -> transposed f16, emb -> normalized f16
  k_cvt_x  <<<dim3(8192), dim3(256), 0, stream>>>(x, xf);     // 8192*256*4 = 8388608
  k_cvt_w  <<<dim3(NSIX), dim3(256), 0, stream>>>(W, Wt);
  k_norm_emb<<<dim3(1024), dim3(256), 0, stream>>>(emb, embf);

  // proj GEMM (WMMA f16): 2048 waves, 8 per block
  k_proj<<<dim3(256), dim3(256), 0, stream>>>(xf, Wt, bia, projf);

  // fused router: 6 heads x 512 row-blocks, 16 waves, 131,328 B dynamic LDS
  k_router<<<dim3(512, 6), dim3(512), 16 * LDSTR * 4, stream>>>(projf, embf, out);

  (void)in_sizes; (void)n_in; (void)out_size; (void)ws_size;
}
